// ShapeDeformCardiacCircleNet_47966194762184
// MI455X (gfx1250) — matrix operand order repaired and checked by reading, not compiled, not run on hardware
//
#include <hip/hip_runtime.h>
#include <hip/hip_bf16.h>

typedef __attribute__((ext_vector_type(16))) _Float16 v16h;
typedef __attribute__((ext_vector_type(8)))  _Float16 v8h;
typedef __attribute__((ext_vector_type(8)))  float    v8f;

#define IMGSZ 128
#define NB    16

static inline int cdiv(long long a, long long b) { return (int)((a + b - 1) / b); }

__device__ __forceinline__ void wait_asynccnt0()
{
#if __has_builtin(__builtin_amdgcn_s_wait_asynccnt)
  __builtin_amdgcn_s_wait_asynccnt(0);
#else
  asm volatile("s_wait_asynccnt 0" ::: "memory");
#endif
}

// ---------------------------------------------------------------------------
// f16 packing helpers
// ---------------------------------------------------------------------------
__global__ void pack_f16_k(const float* __restrict__ src, _Float16* __restrict__ dst,
                           int rows, int K, int K32, int rows16)
{
  size_t tot = (size_t)rows16 * K32;
  size_t i = blockIdx.x * (size_t)blockDim.x + threadIdx.x;
  if (i >= tot) return;
  int r = (int)(i / K32);
  int k = (int)(i - (size_t)r * K32);
  float v = (r < rows && k < K) ? src[(size_t)r * K + k] : 0.0f;
  dst[i] = (_Float16)v;
}

__global__ void cvt_f16_k(const float* __restrict__ src, _Float16* __restrict__ dst, size_t n)
{
  size_t i = blockIdx.x * (size_t)blockDim.x + threadIdx.x;
  if (i < n) dst[i] = (_Float16)src[i];
}

// ---------------------------------------------------------------------------
// Implicit-GEMM convolution on WMMA f16->f32, templated on kernel size.
//   M = Cout, N = B*OH*OW, K = Cin*KH*KW (padded to K32 on the weight side).
//   All 4 waves of a block share one N-tile, so the im2col B fragment is
//   gathered cooperatively into LDS ONCE per block (4 halves per thread,
//   double buffered), then each wave reads it back with two ds_load_b128.
//   Weights are pre-packed f16 [Cout16 x K32] -> two b128 loads per K-step.
// ---------------------------------------------------------------------------
template <int KH, int KW>
__global__ void __launch_bounds__(128) conv2d_wmma_t(
    const _Float16* __restrict__ in, const _Float16* __restrict__ wgt,
    const float* __restrict__ bias, float* __restrict__ out,
    int B, int Cin, int H, int W, int Cout, int OH, int OW,
    int stride, int pad, int K, int K32)
{
  __shared__ _Float16 bstage[2][512];   // 32 lanes x 16 halves, double buffered

  const int t     = threadIdx.x;
  const int lane  = t & 31;
  const int wave  = t >> 5;
  const int tileM = blockIdx.y * 4 + wave;
  const int Cout16 = (Cout + 15) & ~15;
  int m_base = tileM * 16;
  const bool mvalid = (m_base < Cout16);
  if (!mvalid) m_base = 0;              // clamp; stores are guarded below

  const int tileN = blockIdx.x;
  const int N      = B * OH * OW;
  const int n_base = tileN * 16;
  const int hs   = lane >> 4;
  const int l15  = lane & 15;
  const int OHW  = OH * OW;

  // ---- gather role: thread t fills bstage[buf][t*4 .. t*4+3] ------------
  const int tl  = t >> 2;               // target lane 0..31
  const int hb  = (t & 3) * 4;          // target half base 0,4,8,12
  const int ths = tl >> 4;
  const int tn  = n_base + (tl & 15);
  const bool tokN = (tn < N);
  int tnb = 0, tny = 0, tnx = 0;
  if (tokN) {
    tnb = tn / OHW;
    int r = tn - tnb * OHW;
    tny = r / OW;
    tnx = r - tny * OW;
  }
  const int tyb = tny * stride - pad;
  const int txb = tnx * stride - pad;
  const size_t tinb = (size_t)tnb * Cin * H * W;

  auto stage = [&](int buf, int k0) {
#pragma unroll
    for (int k = 0; k < 4; ++k) {
      int h  = hb + k;
      int kb = k0 + ths * 8 + h + (h >= 8 ? 8 : 0);
      int ci = kb / (KH * KW);          // compile-time divisors
      int rr = kb - ci * (KH * KW);
      int ky = rr / KW;
      int kx = rr - ky * KW;
      int iy = tyb + ky;
      int ix = txb + kx;
      bool ok = tokN & (kb < K) & ((unsigned)iy < (unsigned)H) &
                ((unsigned)ix < (unsigned)W);
      size_t idx = ok ? (tinb + ((size_t)ci * H + iy) * W + ix) : 0;
      _Float16 v = in[idx];
      bstage[buf][tl * 16 + h] = ok ? v : (_Float16)0;
    }
  };

  const _Float16* wrow = wgt + (size_t)(m_base + l15) * K32;

  stage(0, 0);
  __syncthreads();

  v8f acc = {};
  int cur = 0;
  for (int k0 = 0; k0 < K32; k0 += 32) {
    __builtin_prefetch(wrow + k0 + 128, 0, 0);
    // ---- B fragment from LDS (shared by all 4 waves)
    v8h b0 = *(const v8h*)(&bstage[cur][lane * 16]);
    v8h b1 = *(const v8h*)(&bstage[cur][lane * 16 + 8]);
    // ---- prefetch next B tile into the other buffer
    if (k0 + 32 < K32) stage(cur ^ 1, k0 + 32);
    // ---- A fragment: two unconditional b128 vector loads (padded)
    v8h a0 = *(const v8h*)(wrow + k0 + hs * 8);
    v8h a1 = *(const v8h*)(wrow + k0 + 16 + hs * 8);
    v16h a, bf;
#pragma unroll
    for (int h = 0; h < 8; ++h) {
      a[h] = a0[h]; a[h + 8] = a1[h];
      bf[h] = b0[h]; bf[h + 8] = b1[h];
    }
    acc = __builtin_amdgcn_wmma_f32_16x16x32_f16(false, a, false, bf,
                                                 (short)0, acc, false, false);
    __syncthreads();
    cur ^= 1;
  }

  if (mvalid) {
#pragma unroll
    for (int r = 0; r < 8; ++r) {
      int m = m_base + r + (hs ? 8 : 0);
      int n = n_base + l15;
      if (m < Cout && n < N) {
        int b2 = n / OHW;
        int rr = n - b2 * OHW;
        int oy = rr / OW;
        int ox = rr - oy * OW;
        out[(((size_t)b2 * Cout + m) * OH + oy) * OW + ox] = acc[r] + bias[m];
      }
    }
  }
}

// ---------------------------------------------------------------------------
// Fully-connected GEMM on WMMA: x [16 x K32] f16 (padded), w [Cout16 x K32]
// f16 (padded). The x tile is staged into a PER-WAVE LDS buffer with
// GLOBAL_LOAD_ASYNC_TO_LDS_B128 (ASYNCcnt pipeline), then read back as the
// B fragment with two ds_load_b128. Per-wave buffers -> no barriers at all;
// each wave waits only on its own ASYNCcnt.
// ---------------------------------------------------------------------------
__global__ void __launch_bounds__(128) linear_wmma(
    const _Float16* __restrict__ x, const _Float16* __restrict__ w,
    const float* __restrict__ bias, float* __restrict__ out,
    int Cout, int K32)
{
  __shared__ _Float16 xs[4][512];       // per-wave 16 rows x 32 halves

  const int lane  = threadIdx.x & 31;
  const int wave  = threadIdx.x >> 5;
  const int tileM = blockIdx.x * 4 + wave;
  const int Cout16 = (Cout + 15) & ~15;
  const int m_base = tileM * 16;
  if (m_base >= Cout16) return;

  const int hs  = lane >> 4;
  const int l15 = lane & 15;
  const _Float16* wrow = w + (size_t)(m_base + l15) * K32;

  // async staging role: lane covers row = lane>>1, halves colh..colh+15
  const int row  = lane >> 1;
  const int colh = (lane & 1) * 16;
  const unsigned ldsb = (unsigned)(uintptr_t)(&xs[wave][0]);
  const unsigned dsta = ldsb + (unsigned)(row * 32 + colh) * 2;

  v8f acc = {};
#pragma unroll 2
  for (int k0 = 0; k0 < K32; k0 += 32) {
    // ---- async global -> LDS staging of the x tile (16 x 32 halves)
    const _Float16* src = x + (size_t)row * K32 + k0 + colh;
    asm volatile("global_load_async_to_lds_b128 %0, %1, off"
                 :: "v"(dsta), "v"(src) : "memory");
    asm volatile("global_load_async_to_lds_b128 %0, %1, off"
                 :: "v"(dsta + 16u), "v"(src + 8) : "memory");
    // ---- A fragment: direct b128 vector loads (padded)
    v8h a0 = *(const v8h*)(wrow + k0 + hs * 8);
    v8h a1 = *(const v8h*)(wrow + k0 + 16 + hs * 8);
    // ---- wait for this wave's async copies, then read B fragment from LDS
    wait_asynccnt0();
    const _Float16* xt = &xs[wave][l15 * 32 + hs * 8];
    v8h b0 = *(const v8h*)(xt);
    v8h b1 = *(const v8h*)(xt + 16);
    v16h a, bf;
#pragma unroll
    for (int h = 0; h < 8; ++h) {
      a[h] = a0[h]; a[h + 8] = a1[h];
      bf[h] = b0[h]; bf[h + 8] = b1[h];
    }
    acc = __builtin_amdgcn_wmma_f32_16x16x32_f16(false, a, false, bf,
                                                 (short)0, acc, false, false);
  }

#pragma unroll
  for (int r = 0; r < 8; ++r) {
    int m = m_base + r + (hs ? 8 : 0);
    int n = l15;
    if (m < Cout && n < NB)
      out[(size_t)n * Cout + m] = acc[r] + bias[m];
  }
}

// ---------------------------------------------------------------------------
// Elementwise / reduction helpers
// ---------------------------------------------------------------------------
__global__ void norm_tensor_k(const float* __restrict__ x, float* __restrict__ y, int n)
{
  int b = blockIdx.x, tid = threadIdx.x;
  const float* p = x + (size_t)b * n;
  float mn = 3.4e38f, mx = -3.4e38f;
  for (int i = tid; i < n; i += 256) { float v = p[i]; mn = fminf(mn, v); mx = fmaxf(mx, v); }
  __shared__ float smn[256], smx[256];
  smn[tid] = mn; smx[tid] = mx; __syncthreads();
  for (int o = 128; o > 0; o >>= 1) {
    if (tid < o) { smn[tid] = fminf(smn[tid], smn[tid + o]); smx[tid] = fmaxf(smx[tid], smx[tid + o]); }
    __syncthreads();
  }
  float lo = smn[0], hi = smx[0];
  float inv = 1.0f / (hi - lo + 0.01f);
  float* q = y + (size_t)b * n;
  for (int i = tid; i < n; i += 256) q[i] = (p[i] - lo) * inv;
}

__global__ void bn2d_stats(const float* __restrict__ x, float* __restrict__ mean,
                           float* __restrict__ inv, int B, int C, int HW)
{
  int c = blockIdx.x, tid = threadIdx.x;
  float s = 0.f, s2 = 0.f;
  int n = B * HW;
  for (int i = tid; i < n; i += 256) {
    int b = i / HW, r = i - b * HW;
    float v = x[((size_t)b * C + c) * HW + r];
    s += v; s2 += v * v;
  }
  __shared__ float sh[256], sh2[256];
  sh[tid] = s; sh2[tid] = s2; __syncthreads();
  for (int o = 128; o > 0; o >>= 1) {
    if (tid < o) { sh[tid] += sh[tid + o]; sh2[tid] += sh2[tid + o]; }
    __syncthreads();
  }
  if (tid == 0) {
    float m = sh[0] / n;
    float v = sh2[0] / n - m * m;
    mean[c] = m; inv[c] = rsqrtf(v + 1e-5f);
  }
}

__global__ void bn2d_apply(float* __restrict__ x, const float* __restrict__ mean,
                           const float* __restrict__ inv, const float* __restrict__ g,
                           const float* __restrict__ be, int B, int C, int HW, int relu)
{
  size_t i = blockIdx.x * (size_t)blockDim.x + threadIdx.x;
  size_t tot = (size_t)B * C * HW;
  if (i >= tot) return;
  int c = (int)((i / HW) % C);
  float v = (x[i] - mean[c]) * inv[c] * g[c] + be[c];
  if (relu) v = fmaxf(v, 0.f);
  x[i] = v;
}

__global__ void bn1d_relu(float* __restrict__ x, const float* __restrict__ g,
                          const float* __restrict__ be, int B, int F)
{
  int f = blockIdx.x * blockDim.x + threadIdx.x;
  if (f >= F) return;
  float s = 0.f, s2 = 0.f;
  for (int b = 0; b < B; ++b) { float v = x[b * F + f]; s += v; s2 += v * v; }
  float m = s / B, var = s2 / B - m * m, inv = rsqrtf(var + 1e-5f);
  for (int b = 0; b < B; ++b) {
    float v = (x[b * F + f] - m) * inv * g[f] + be[f];
    x[b * F + f] = fmaxf(v, 0.f);
  }
}

__global__ void maxpool2_k(const float* __restrict__ x, float* __restrict__ y,
                           int BC, int H, int W)
{
  int OH = H >> 1, OW = W >> 1;
  size_t tot = (size_t)BC * OH * OW;
  size_t i = blockIdx.x * (size_t)blockDim.x + threadIdx.x;
  if (i >= tot) return;
  int ox = (int)(i % OW);
  int oy = (int)((i / OW) % OH);
  int bc = (int)(i / ((size_t)OW * OH));
  const float* p = x + (size_t)bc * H * W;
  float a = p[(2 * oy) * W + 2 * ox];
  float b = p[(2 * oy) * W + 2 * ox + 1];
  float c = p[(2 * oy + 1) * W + 2 * ox];
  float d = p[(2 * oy + 1) * W + 2 * ox + 1];
  y[i] = fmaxf(fmaxf(a, b), fmaxf(c, d));
}

__global__ void head_k(const float* __restrict__ x, const float* __restrict__ w,
                       const float* __restrict__ bias, float* __restrict__ out,
                       int B, int C, int K, int act)
{
  int i = blockIdx.x * blockDim.x + threadIdx.x;
  if (i >= B * C) return;
  int b = i / C, c = i - b * C;
  float s = bias[c];
  for (int k = 0; k < K; ++k) s += x[b * K + k] * w[c * K + k];
  if (act == 1) s = tanhf(s);
  else if (act == 2) s = 1.f / (1.f + expf(-s));
  out[i] = s;
}

// ---------------------------------------------------------------------------
// Circle generation + SDF mask rasterization
// ---------------------------------------------------------------------------
__global__ void circles_k(const float* __restrict__ p1, const float* __restrict__ p2,
                          float* __restrict__ n0, float* __restrict__ n1,
                          float* __restrict__ n2, int B)
{
  int b = threadIdx.x;
  if (b >= B) return;
  float ctrx = 0.5f * p1[b * 2 + 0], ctry = 0.5f * p1[b * 2 + 1];
  float r0 = 0.25f * p2[b * 5 + 0];
  float r1 = r0 * (1.f + p2[b * 5 + 1]);
  float r2 = 0.2f * (p2[b * 5 + 3] + p2[b * 5 + 4]);
  float c2x = ctrx - (r1 + 0.25f * p2[b * 5 + 2]);
  float c2y = ctry;
  for (int n = 0; n < 32; ++n) {
    float t = 6.283185307179586f * n / 32.f;
    float ux = cosf(t), uy = sinf(t);
    n0[(b * 33 + n) * 2 + 0] = ctrx + r0 * ux;
    n0[(b * 33 + n) * 2 + 1] = ctry + r0 * uy;
    n1[(b * 33 + n) * 2 + 0] = ctrx + r1 * ux;
    n1[(b * 33 + n) * 2 + 1] = ctry + r1 * uy;
    n2[(b * 33 + n) * 2 + 0] = c2x + r2 * ux;
    n2[(b * 33 + n) * 2 + 1] = c2y + r2 * uy;
  }
  n0[(b * 33 + 32) * 2 + 0] = ctrx; n0[(b * 33 + 32) * 2 + 1] = ctry;
  n1[(b * 33 + 32) * 2 + 0] = ctrx; n1[(b * 33 + 32) * 2 + 1] = ctry;
  n2[(b * 33 + 32) * 2 + 0] = c2x;  n2[(b * 33 + 32) * 2 + 1] = c2y;
}

__global__ void render_mask_k(const float* __restrict__ nodes, float* __restrict__ mask, int B)
{
  size_t tot = (size_t)B * IMGSZ * IMGSZ;
  size_t i = blockIdx.x * (size_t)blockDim.x + threadIdx.x;
  if (i >= tot) return;
  int x = (int)(i % IMGSZ);
  int y = (int)((i / IMGSZ) % IMGSZ);
  int b = (int)(i / (IMGSZ * IMGSZ));
  float px = ((x + 0.5f) / IMGSZ) * 2.f - 1.f;
  float py = ((y + 0.5f) / IMGSZ) * 2.f - 1.f;
  float cx = nodes[(b * 33 + 32) * 2 + 0];
  float cy = nodes[(b * 33 + 32) * 2 + 1];
  float best = -3.4e38f;
  for (int f = 0; f < 32; ++f) {
    float ax = nodes[(b * 33 + f) * 2 + 0], ay = nodes[(b * 33 + f) * 2 + 1];
    int f2 = (f + 1) & 31;
    float bx = nodes[(b * 33 + f2) * 2 + 0], by = nodes[(b * 33 + f2) * 2 + 1];
    float e01x = bx - ax, e01y = by - ay;
    float e02x = cx - ax, e02y = cy - ay;
    float cr = e01x * e02y - e01y * e02x;
    float sgn = (cr > 0.f) ? 1.f : ((cr < 0.f) ? -1.f : 0.f);
    float d0 = (e01x * (py - ay) - e01y * (px - ax)) *
               rsqrtf(e01x * e01x + e01y * e01y + 1e-12f) * sgn;
    float ex = cx - bx, ey = cy - by;
    float d1 = (ex * (py - by) - ey * (px - bx)) *
               rsqrtf(ex * ex + ey * ey + 1e-12f) * sgn;
    float ex2 = ax - cx, ey2 = ay - cy;
    float d2 = (ex2 * (py - cy) - ey2 * (px - cx)) *
               rsqrtf(ex2 * ex2 + ey2 * ey2 + 1e-12f) * sgn;
    float d = fminf(d0, fminf(d1, d2));
    float sg = 1.f / (1.f + expf(-d * (IMGSZ * 0.5f)));
    best = fmaxf(best, sg);
  }
  mask[i] = best;
}

// ---------------------------------------------------------------------------
// Similarity matrix theta = (R*S*T)[:, :2, :]
// ---------------------------------------------------------------------------
__global__ void similarity_k(const float* __restrict__ ap1, const float* __restrict__ ap2,
                             float* __restrict__ theta, int B)
{
  int b = threadIdx.x;
  if (b >= B) return;
  float th = ap1[b * 3 + 0] * 3.14159265358979f;
  float c = cosf(th), s = sinf(th);
  float tx = ap1[b * 3 + 1], ty = ap1[b * 3 + 2];
  float s0 = ap2[b * 2 + 0], s1 = ap2[b * 2 + 1];
  float R[3][3] = {{c, s, 0.f}, {-s, c, 0.f}, {0.f, 0.f, 1.f}};
  float S[3][3] = {{s0, 0.f, 0.f}, {0.f, s1, 0.f}, {0.f, 0.f, 1.f}};
  float T[3][3] = {{1.f, 0.f, 0.f}, {0.f, 1.f, 0.f}, {tx, ty, 1.f}};
  float RS[3][3];
  for (int i = 0; i < 3; ++i)
    for (int j = 0; j < 3; ++j) {
      float a = 0.f;
      for (int k = 0; k < 3; ++k) a += R[i][k] * S[k][j];
      RS[i][j] = a;
    }
  for (int i = 0; i < 2; ++i)
    for (int j = 0; j < 3; ++j) {
      float a = 0.f;
      for (int k = 0; k < 3; ++k) a += RS[i][k] * T[k][j];
      theta[b * 6 + i * 3 + j] = a;
    }
}

// ---------------------------------------------------------------------------
// Bilinear helpers / warps / resize
// ---------------------------------------------------------------------------
__device__ inline float bilin_zero(const float* __restrict__ p, int H, int W,
                                   float ix, float iy)
{
  float x0f = floorf(ix), y0f = floorf(iy);
  int x0 = (int)x0f, y0 = (int)y0f;
  float wx1 = ix - x0f, wx0 = 1.f - wx1;
  float wy1 = iy - y0f, wy0 = 1.f - wy1;
  float r = 0.f;
  if (x0 >= 0 && x0 < W && y0 >= 0 && y0 < H)         r += p[y0 * W + x0] * wx0 * wy0;
  if (x0 + 1 >= 0 && x0 + 1 < W && y0 >= 0 && y0 < H) r += p[y0 * W + x0 + 1] * wx1 * wy0;
  if (x0 >= 0 && x0 < W && y0 + 1 >= 0 && y0 + 1 < H) r += p[(y0 + 1) * W + x0] * wx0 * wy1;
  if (x0 + 1 >= 0 && x0 + 1 < W && y0 + 1 >= 0 && y0 + 1 < H)
    r += p[(y0 + 1) * W + x0 + 1] * wx1 * wy1;
  return r;
}

__global__ void affine_warp_k(const float* __restrict__ img, const float* __restrict__ theta,
                              float* __restrict__ out, int B, int C, int H, int W)
{
  size_t tot = (size_t)B * C * H * W;
  size_t i = blockIdx.x * (size_t)blockDim.x + threadIdx.x;
  if (i >= tot) return;
  int x = (int)(i % W);
  int y = (int)((i / W) % H);
  int c = (int)((i / ((size_t)W * H)) % C);
  int b = (int)(i / ((size_t)W * H * C));
  float X = (2.f * x + 1.f) / W - 1.f;
  float Y = (2.f * y + 1.f) / H - 1.f;
  const float* t = theta + b * 6;
  float gx = t[0] * X + t[1] * Y + t[2];
  float gy = t[3] * X + t[4] * Y + t[5];
  float ix = ((gx + 1.f) * W - 1.f) * 0.5f;
  float iy = ((gy + 1.f) * H - 1.f) * 0.5f;
  out[i] = bilin_zero(img + ((size_t)b * C + c) * H * W, H, W, ix, iy);
}

__global__ void deform_warp_k(const float* __restrict__ img, const float* __restrict__ flow,
                              float* __restrict__ out, int B, int C, int H, int W)
{
  size_t tot = (size_t)B * C * H * W;
  size_t i = blockIdx.x * (size_t)blockDim.x + threadIdx.x;
  if (i >= tot) return;
  int x = (int)(i % W);
  int y = (int)((i / W) % H);
  int c = (int)((i / ((size_t)W * H)) % C);
  int b = (int)(i / ((size_t)W * H * C));
  float fy = flow[(((size_t)b * 2 + 0) * H + y) * W + x];
  float fx = flow[(((size_t)b * 2 + 1) * H + y) * W + x];
  out[i] = bilin_zero(img + ((size_t)b * C + c) * H * W, H, W, x + fx, y + fy);
}

__global__ void vecint_step_k(const float* __restrict__ v, float* __restrict__ out,
                              int B, int H, int W)
{
  size_t tot = (size_t)B * 2 * H * W;
  size_t i = blockIdx.x * (size_t)blockDim.x + threadIdx.x;
  if (i >= tot) return;
  int x = (int)(i % W);
  int y = (int)((i / W) % H);
  int c = (int)((i / ((size_t)W * H)) % 2);
  int b = (int)(i / ((size_t)W * H * 2));
  float fy = v[(((size_t)b * 2 + 0) * H + y) * W + x];
  float fx = v[(((size_t)b * 2 + 1) * H + y) * W + x];
  out[i] = v[i] + bilin_zero(v + ((size_t)b * 2 + c) * H * W, H, W, x + fx, y + fy);
}

__global__ void resize_bilinear_k(const float* __restrict__ in, float* __restrict__ out,
                                  int B, int C, int IH, int IW, int OH, int OW)
{
  size_t tot = (size_t)B * C * OH * OW;
  size_t i = blockIdx.x * (size_t)blockDim.x + threadIdx.x;
  if (i >= tot) return;
  int ox = (int)(i % OW);
  int oy = (int)((i / OW) % OH);
  int c  = (int)((i / ((size_t)OW * OH)) % C);
  int b  = (int)(i / ((size_t)OW * OH * C));
  float sy = (oy + 0.5f) * IH / (float)OH - 0.5f;
  float sx = (ox + 0.5f) * IW / (float)OW - 0.5f;
  float y0f = floorf(sy), x0f = floorf(sx);
  float wy1 = sy - y0f, wx1 = sx - x0f;
  int y0 = min(max((int)y0f, 0), IH - 1);
  int y1 = min(max((int)y0f + 1, 0), IH - 1);
  int x0 = min(max((int)x0f, 0), IW - 1);
  int x1 = min(max((int)x0f + 1, 0), IW - 1);
  const float* p = in + ((size_t)b * C + c) * IH * IW;
  out[i] = (1.f - wy1) * ((1.f - wx1) * p[y0 * IW + x0] + wx1 * p[y0 * IW + x1]) +
           wy1 * ((1.f - wx1) * p[y1 * IW + x0] + wx1 * p[y1 * IW + x1]);
}

// ---------------------------------------------------------------------------
// Concats / slices / copies / scale
// ---------------------------------------------------------------------------
__global__ void concat2_k(const float* __restrict__ a, int Ca,
                          const float* __restrict__ b, int Cb,
                          float* __restrict__ out, int B, int HW)
{
  int C = Ca + Cb;
  size_t tot = (size_t)B * C * HW;
  size_t i = blockIdx.x * (size_t)blockDim.x + threadIdx.x;
  if (i >= tot) return;
  int hw = (int)(i % HW);
  int c  = (int)((i / HW) % C);
  int bb = (int)(i / ((size_t)HW * C));
  out[i] = (c < Ca) ? a[((size_t)bb * Ca + c) * HW + hw]
                    : b[((size_t)bb * Cb + (c - Ca)) * HW + hw];
}

__global__ void cat1111_k(const float* __restrict__ a, const float* __restrict__ b,
                          const float* __restrict__ c, const float* __restrict__ d,
                          float* __restrict__ out, int B, int HW)
{
  size_t tot = (size_t)B * 4 * HW;
  size_t i = blockIdx.x * (size_t)blockDim.x + threadIdx.x;
  if (i >= tot) return;
  int hw = (int)(i % HW);
  int ch = (int)((i / HW) % 4);
  int bb = (int)(i / ((size_t)HW * 4));
  const float* src = (ch == 0) ? a : (ch == 1) ? b : (ch == 2) ? c : d;
  out[i] = src[(size_t)bb * HW + hw];
}

__global__ void cat111_k(const float* __restrict__ a, const float* __restrict__ b,
                         const float* __restrict__ c, float* __restrict__ out,
                         int B, int HW)
{
  size_t tot = (size_t)B * 3 * HW;
  size_t i = blockIdx.x * (size_t)blockDim.x + threadIdx.x;
  if (i >= tot) return;
  int hw = (int)(i % HW);
  int ch = (int)((i / HW) % 3);
  int bb = (int)(i / ((size_t)HW * 3));
  const float* src = (ch == 0) ? a : (ch == 1) ? b : c;
  out[i] = src[(size_t)bb * HW + hw];
}

__global__ void slice_ch_k(const float* __restrict__ src, float* __restrict__ dst,
                           int B, int C, int HW, int ch)
{
  size_t tot = (size_t)B * HW;
  size_t i = blockIdx.x * (size_t)blockDim.x + threadIdx.x;
  if (i >= tot) return;
  int bb = (int)(i / HW);
  int hw = (int)(i % HW);
  dst[i] = src[((size_t)bb * C + ch) * HW + hw];
}

__global__ void copy_k(const float* __restrict__ src, float* __restrict__ dst, size_t n)
{
  size_t i = blockIdx.x * (size_t)blockDim.x + threadIdx.x;
  if (i < n) dst[i] = src[i];
}

__global__ void scale_k(const float* __restrict__ src, float* __restrict__ dst,
                        float s, size_t n)
{
  size_t i = blockIdx.x * (size_t)blockDim.x + threadIdx.x;
  if (i < n) dst[i] = src[i] * s;
}

// ---------------------------------------------------------------------------
// Host-side orchestration
// ---------------------------------------------------------------------------
static void launch_conv(const float* in, const float* w, const float* b, float* out,
                        _Float16* wh, _Float16* inh,
                        int B, int Cin, int H, int W, int Cout, int k, int stride,
                        int pad, hipStream_t s)
{
  int K = Cin * k * k;
  int K32 = (K + 31) & ~31;
  int Cout16 = (Cout + 15) & ~15;

  // pack weights f16 padded [Cout16 x K32]
  pack_f16_k<<<cdiv((long long)Cout16 * K32, 256), 256, 0, s>>>(w, wh, Cout, K, K32, Cout16);

  if (H == 1 && W == 1 && k == 1) {
    // FC path: pack input rows padded too
    pack_f16_k<<<cdiv((long long)16 * K32, 256), 256, 0, s>>>(in, inh, B, K, K32, 16);
    int tilesM = Cout16 / 16;
    linear_wmma<<<cdiv(tilesM, 4), 128, 0, s>>>(inh, wh, b, out, Cout, K32);
    return;
  }

  size_t n = (size_t)B * Cin * H * W;
  cvt_f16_k<<<cdiv(n, 256), 256, 0, s>>>(in, inh, n);

  int OH = (H + stride - 1) / stride;
  int OW = (W + stride - 1) / stride;
  long long N = (long long)B * OH * OW;
  dim3 grid(cdiv(N, 16), cdiv(Cout16 / 16, 4));
  if (k == 5)
    conv2d_wmma_t<5, 5><<<grid, 128, 0, s>>>(inh, wh, b, out, B, Cin, H, W, Cout,
                                             OH, OW, stride, pad, K, K32);
  else if (k == 3)
    conv2d_wmma_t<3, 3><<<grid, 128, 0, s>>>(inh, wh, b, out, B, Cin, H, W, Cout,
                                             OH, OW, stride, pad, K, K32);
  else
    conv2d_wmma_t<1, 1><<<grid, 128, 0, s>>>(inh, wh, b, out, B, Cin, H, W, Cout,
                                             OH, OW, stride, pad, K, K32);
}

static void conv_bn(const float* in, void* const* d_in, int pbase, float* out,
                    float* stat, _Float16* wh, _Float16* inh,
                    int B, int Cin, int H, int W, int Cout, int k,
                    int stride, int pad, int relu, hipStream_t s)
{
  const float* w  = (const float*)d_in[pbase + 0];
  const float* b  = (const float*)d_in[pbase + 1];
  const float* g  = (const float*)d_in[pbase + 2];
  const float* be = (const float*)d_in[pbase + 3];
  int OH = (H + stride - 1) / stride;
  int OW = (W + stride - 1) / stride;
  launch_conv(in, w, b, out, wh, inh, B, Cin, H, W, Cout, k, stride, pad, s);
  bn2d_stats<<<Cout, 256, 0, s>>>(out, stat, stat + 1024, B, Cout, OH * OW);
  long long tot = (long long)B * Cout * OH * OW;
  bn2d_apply<<<cdiv(tot, 256), 256, 0, s>>>(out, stat, stat + 1024, g, be,
                                            B, Cout, OH * OW, relu);
}

static void run_encoder(const float* x, int Cin, int base, void* const* d_in,
                        float* e1, float* e2, float* e3, float* e4, float* stat,
                        _Float16* wh, _Float16* inh, hipStream_t s)
{
  conv_bn(x,  d_in, base + 0,  e1, stat, wh, inh, NB, Cin, 128, 128,  64, 5, 2, 1, 1, s);
  conv_bn(e1, d_in, base + 4,  e2, stat, wh, inh, NB,  64,  64,  64, 128, 5, 2, 1, 1, s);
  conv_bn(e2, d_in, base + 8,  e3, stat, wh, inh, NB, 128,  32,  32, 256, 5, 2, 1, 1, s);
  conv_bn(e3, d_in, base + 12, e4, stat, wh, inh, NB, 256,  16,  16, 512, 5, 2, 1, 1, s);
}

static void run_regressor(const float* f4, int base, void* const* d_in,
                          float* rc1, float* rpool, float* rc2, float* rf1,
                          float* rf2, float* stat, _Float16* wh, _Float16* inh,
                          hipStream_t s)
{
  conv_bn(f4, d_in, base + 0, rc1, stat, wh, inh, NB, 512, 8, 8, 256, 5, 1, 2, 1, s);
  maxpool2_k<<<cdiv((long long)NB * 256 * 4 * 4, 256), 256, 0, s>>>(rc1, rpool, NB * 256, 8, 8);
  conv_bn(rpool, d_in, base + 4, rc2, stat, wh, inh, NB, 256, 4, 4, 128, 1, 1, 0, 1, s);
  // f1: 2048 -> 400 (FC path)
  launch_conv(rc2, (const float*)d_in[base + 8], (const float*)d_in[base + 9], rf1,
              wh, inh, NB, 2048, 1, 1, 400, 1, 1, 0, s);
  bn1d_relu<<<cdiv(400, 128), 128, 0, s>>>(rf1, (const float*)d_in[base + 10],
                                           (const float*)d_in[base + 11], NB, 400);
  // f2: 400 -> 200 (FC path)
  launch_conv(rf1, (const float*)d_in[base + 12], (const float*)d_in[base + 13], rf2,
              wh, inh, NB, 400, 1, 1, 200, 1, 1, 0, s);
  bn1d_relu<<<cdiv(200, 128), 128, 0, s>>>(rf2, (const float*)d_in[base + 14],
                                           (const float*)d_in[base + 15], NB, 200);
}

extern "C" void kernel_launch(void* const* d_in, const int* in_sizes, int n_in,
                              void* d_out, int out_size, void* d_ws, size_t ws_size,
                              hipStream_t stream)
{
  (void)in_sizes; (void)n_in; (void)out_size; (void)ws_size;
  // Input leaf order (depth-first insertion order of setup_inputs dict):
  // 0: img
  // 1..16  shape_enc; 17..32 shape_reg; 33,34 shape_end1; 35,36 shape_end2
  // 37..52 affine_enc; 53..68 affine_reg; 69,70 affine_end1; 71,72 affine_end2
  // 73..88 deform_enc; 89..100 dec; 101,102 flow
  const int SHAPE_ENC = 1, SHAPE_REG = 17, SHAPE_E1 = 33, SHAPE_E2 = 35;
  const int AFF_ENC = 37, AFF_REG = 53, AFF_E1 = 69, AFF_E2 = 71;
  const int DEF_ENC = 73, DEC = 89, FLOWP = 101;

  const float* img = (const float*)d_in[0];
  float* ws = (float*)d_ws;
  size_t off = 0;
  auto A = [&](size_t n) { float* p = ws + off; off += (n + 63) & ~(size_t)63; return p; };

  float* stat   = A(2048);
  float* xn     = A(262144);
  float* e1     = A(4194304);
  float* e2     = A(2097152);
  float* e3     = A(1048576);
  float* e4     = A(524288);
  float* rc1    = A(262144);
  float* rpool  = A(65536);
  float* rc2    = A(32768);
  float* rf1    = A(6400);
  float* rf2    = A(3200);
  float* p1     = A(64);
  float* p2     = A(128);
  float* ap1    = A(64);
  float* ap2    = A(64);
  float* theta  = A(128);
  float* nod0   = A(1056);
  float* nod1   = A(1056);
  float* nod2   = A(1056);
  float* m0     = A(262144);
  float* m1     = A(262144);
  float* m2     = A(262144);
  float* cat4   = A(1048576);
  float* initm  = A(786432);
  float* affm   = A(786432);
  float* defm   = A(786432);
  float* upA    = A(8388608);    // up-sample arena (largest stage)
  float* catA   = A(12582912);   // concat arena (largest stage)
  float* d3     = A(1048576);
  float* d2b    = A(2097152);
  float* d1b    = A(8388608);
  float* fl64   = A(131072);
  float* preint = A(524288);
  float* vA     = A(524288);
  float* vB     = A(524288);
  _Float16* wh  = (_Float16*)A(2600000);   // 5.2M halves: padded f16 weights
  _Float16* inh = (_Float16*)A(6400000);   // 12.8M halves: f16 activations

  const int HW = IMGSZ * IMGSZ;

  // ---- normalize input -------------------------------------------------
  norm_tensor_k<<<NB, 256, 0, stream>>>(img, xn, HW);

  // ---- shape branch ----------------------------------------------------
  run_encoder(xn, 1, SHAPE_ENC, d_in, e1, e2, e3, e4, stat, wh, inh, stream);
  run_regressor(e4, SHAPE_REG, d_in, rc1, rpool, rc2, rf1, rf2, stat, wh, inh, stream);
  head_k<<<1, 64, 0, stream>>>(rf2, (const float*)d_in[SHAPE_E1], (const float*)d_in[SHAPE_E1 + 1], p1, NB, 2, 200, 1);
  head_k<<<1, 128, 0, stream>>>(rf2, (const float*)d_in[SHAPE_E2], (const float*)d_in[SHAPE_E2 + 1], p2, NB, 5, 200, 2);
  circles_k<<<1, 32, 0, stream>>>(p1, p2, nod0, nod1, nod2, NB);
  render_mask_k<<<cdiv((long long)NB * HW, 256), 256, 0, stream>>>(nod0, m0, NB);
  render_mask_k<<<cdiv((long long)NB * HW, 256), 256, 0, stream>>>(nod1, m1, NB);
  render_mask_k<<<cdiv((long long)NB * HW, 256), 256, 0, stream>>>(nod2, m2, NB);

  // ---- affine branch ---------------------------------------------------
  cat1111_k<<<cdiv((long long)NB * 4 * HW, 256), 256, 0, stream>>>(xn, m0, m1, m2, cat4, NB, HW);
  run_encoder(cat4, 4, AFF_ENC, d_in, e1, e2, e3, e4, stat, wh, inh, stream);
  run_regressor(e4, AFF_REG, d_in, rc1, rpool, rc2, rf1, rf2, stat, wh, inh, stream);
  head_k<<<1, 64, 0, stream>>>(rf2, (const float*)d_in[AFF_E1], (const float*)d_in[AFF_E1 + 1], ap1, NB, 3, 200, 1);
  head_k<<<1, 64, 0, stream>>>(rf2, (const float*)d_in[AFF_E2], (const float*)d_in[AFF_E2 + 1], ap2, NB, 2, 200, 0);
  similarity_k<<<1, 32, 0, stream>>>(ap1, ap2, theta, NB);
  cat111_k<<<cdiv((long long)NB * 3 * HW, 256), 256, 0, stream>>>(m0, m1, m2, initm, NB, HW);
  affine_warp_k<<<cdiv((long long)NB * 3 * HW, 256), 256, 0, stream>>>(initm, theta, affm, NB, 3, IMGSZ, IMGSZ);

  // ---- deform branch: encoder + U-Net decoder --------------------------
  concat2_k<<<cdiv((long long)NB * 4 * HW, 256), 256, 0, stream>>>(xn, 1, affm, 3, cat4, NB, HW);
  run_encoder(cat4, 4, DEF_ENC, d_in, e1, e2, e3, e4, stat, wh, inh, stream);

  // stage 1: f4 (512,8,8) -> 16x16, cat f3 (256) -> conv 768->256
  resize_bilinear_k<<<cdiv((long long)NB * 512 * 256, 256), 256, 0, stream>>>(e4, upA, NB, 512, 8, 8, 16, 16);
  concat2_k<<<cdiv((long long)NB * 768 * 256, 256), 256, 0, stream>>>(upA, 512, e3, 256, catA, NB, 256);
  conv_bn(catA, d_in, DEC + 0, d3, stat, wh, inh, NB, 768, 16, 16, 256, 5, 1, 2, 1, stream);

  // stage 2: d3 (256,16,16) -> 32x32, cat f2 (128) -> conv 384->128
  resize_bilinear_k<<<cdiv((long long)NB * 256 * 1024, 256), 256, 0, stream>>>(d3, upA, NB, 256, 16, 16, 32, 32);
  concat2_k<<<cdiv((long long)NB * 384 * 1024, 256), 256, 0, stream>>>(upA, 256, e2, 128, catA, NB, 1024);
  conv_bn(catA, d_in, DEC + 4, d2b, stat, wh, inh, NB, 384, 32, 32, 128, 5, 1, 2, 1, stream);

  // stage 3: d2 (128,32,32) -> 64x64, cat f1 (64) -> conv 192->128 (no relu)
  resize_bilinear_k<<<cdiv((long long)NB * 128 * 4096, 256), 256, 0, stream>>>(d2b, upA, NB, 128, 32, 32, 64, 64);
  concat2_k<<<cdiv((long long)NB * 192 * 4096, 256), 256, 0, stream>>>(upA, 128, e1, 64, catA, NB, 4096);
  conv_bn(catA, d_in, DEC + 8, d1b, stat, wh, inh, NB, 192, 64, 64, 128, 5, 1, 2, 0, stream);

  // flow head: 128 -> 2, 3x3, then upsample to 128x128
  launch_conv(d1b, (const float*)d_in[FLOWP], (const float*)d_in[FLOWP + 1], fl64,
              wh, inh, NB, 128, 64, 64, 2, 3, 1, 1, stream);
  resize_bilinear_k<<<cdiv((long long)NB * 2 * HW, 256), 256, 0, stream>>>(fl64, preint, NB, 2, 64, 64, IMGSZ, IMGSZ);

  // ---- vecint: scaling & squaring (7 steps) ----------------------------
  size_t fn = (size_t)NB * 2 * HW;
  scale_k<<<cdiv(fn, 256), 256, 0, stream>>>(preint, vA, 1.0f / 128.0f, fn);
  float* vcur = vA; float* vnext = vB;
  for (int it = 0; it < 7; ++it) {
    vecint_step_k<<<cdiv(fn, 256), 256, 0, stream>>>(vcur, vnext, NB, IMGSZ, IMGSZ);
    float* t = vcur; vcur = vnext; vnext = t;
  }
  deform_warp_k<<<cdiv((long long)NB * 3 * HW, 256), 256, 0, stream>>>(affm, vcur, defm, NB, 3, IMGSZ, IMGSZ);

  // ---- assemble outputs ------------------------------------------------
  float* o = (float*)d_out;
  size_t flown = (size_t)NB * 2 * HW;   // 524288
  size_t maskn = (size_t)NB * HW;       // 262144
  copy_k<<<cdiv(flown, 256), 256, 0, stream>>>(vcur, o, flown);   o += flown;
  copy_k<<<cdiv(flown, 256), 256, 0, stream>>>(preint, o, flown); o += flown;
  copy_k<<<cdiv(maskn, 256), 256, 0, stream>>>(m0, o, maskn);     o += maskn;
  copy_k<<<cdiv(maskn, 256), 256, 0, stream>>>(m1, o, maskn);     o += maskn;
  copy_k<<<cdiv(maskn, 256), 256, 0, stream>>>(m2, o, maskn);     o += maskn;
  for (int c = 0; c < 3; ++c) {
    slice_ch_k<<<cdiv(maskn, 256), 256, 0, stream>>>(affm, o, NB, 3, HW, c);
    o += maskn;
  }
  for (int c = 0; c < 3; ++c) {
    slice_ch_k<<<cdiv(maskn, 256), 256, 0, stream>>>(defm, o, NB, 3, HW, c);
    o += maskn;
  }
}
